// GINBranchLayer_79147657331007
// MI455X (gfx1250) — compile-verified, matrix operand order checked by reading, not run on hardware
//
#include <hip/hip_runtime.h>
#include <hip/hip_bf16.h>
#include <stdint.h>

typedef __attribute__((ext_vector_type(16))) _Float16 v16h;
typedef __attribute__((ext_vector_type(8)))  float    v8f;

#define HID 128
#define LDS_STRIDE 136  // 128 + 8 halves padding -> avoids 64-bank conflicts

// ---------------------------------------------------------------------------
// Fragment load from row-major (padded) f16 LDS.
// CDNA5 16-bit A-matrix 16x32 layout (ISA 7.12.2): lane group g = lane>>4,
// half-pair q (VGPR q) holds K = (q<4 ? 2q : 2q+8) + 8g  and K+1.
// u32 word indices {0,1,2,3} and {8,9,10,11} (+4g) -> compiler emits two
// ds_load_b128 per fragment (verified in round-1 disasm).
// ---------------------------------------------------------------------------
__device__ __forceinline__ v16h frag_ld(const _Float16* __restrict__ p, int g) {
  union { v16h v; uint32_t u[8]; } f;
  const int g8 = g << 3;
  #pragma unroll
  for (int q = 0; q < 8; ++q) {
    const int K = (q < 4 ? 2 * q : 2 * q + 8) + g8;
    f.u[q] = *(const uint32_t*)(p + K);   // even half index -> 4B aligned
  }
  return f.v;
}

// ---------------------------------------------------------------------------
// Pack W1..W4 (row-major [K=128][N=128] f32) into transposed f16 [n][k],
// and fold BN params into per-feature scale/shift.
// Grid: 256 blocks x 256 threads == 4*128*128 elements.
// ---------------------------------------------------------------------------
__global__ void pack_weights(const float* __restrict__ W1, const float* __restrict__ W2,
                             const float* __restrict__ W3, const float* __restrict__ W4,
                             const float* __restrict__ g1, const float* __restrict__ be1,
                             const float* __restrict__ mm1, const float* __restrict__ mv1,
                             const float* __restrict__ g2, const float* __restrict__ be2,
                             const float* __restrict__ mm2, const float* __restrict__ mv2,
                             _Float16* __restrict__ wt, float* __restrict__ bn) {
  const int idx = blockIdx.x * 256 + threadIdx.x;
  const float* Ws[4] = {W1, W2, W3, W4};
  const int w   = idx >> 14;
  const int rem = idx & 16383;
  const int n   = rem >> 7;
  const int k   = rem & 127;
  wt[w * 16384 + n * 128 + k] = (_Float16)Ws[w][k * 128 + n];

  if (idx < 128) {
    const float s = g1[idx] * rsqrtf(mv1[idx] + 1e-3f);
    bn[idx]       = s;
    bn[128 + idx] = be1[idx] - mm1[idx] * s;
  } else if (idx < 256) {
    const int j = idx - 128;
    const float s = g2[j] * rsqrtf(mv2[j] + 1e-3f);
    bn[256 + j] = s;
    bn[384 + j] = be2[j] - mm2[j] * s;
  }
}

// ---------------------------------------------------------------------------
// out[row,:] = epilogue( in[row,:] @ W + bias ),  W given as transposed f16.
// epilogue: ReLU, then optional BN (y = y*s + t). If out2 != null the result
// is stored twice (pre-seeds the aggregation accumulator, replacing a full
// d2d copy). Safe even when out2 aliases `in`: each block reads only the 64
// rows it writes, and all staging loads precede the stores in program order.
// Block: 256 threads (8 waves), 64 rows x 128 cols; 16 WMMAs / wave.
// ---------------------------------------------------------------------------
__global__ __launch_bounds__(256)
void gemm128(const float* __restrict__ in, float* __restrict__ out,
             float* __restrict__ out2,
             const uint32_t* __restrict__ wt32, const float* __restrict__ bias,
             const float* __restrict__ bnS, const float* __restrict__ bnT, int Nn) {
  __shared__ _Float16 sA[64 * LDS_STRIDE];
  __shared__ _Float16 sB[128 * LDS_STRIDE];

  const int tid  = threadIdx.x;
  const int row0 = blockIdx.x * 64;

  // ---- stage A: 64 rows of fp32 activations -> f16 LDS (padded stride)
  const float4* in4 = (const float4*)in;
  #pragma unroll
  for (int i = tid; i < 64 * 32; i += 256) {
    const int r  = i >> 5;
    const int c4 = i & 31;
    const int row = row0 + r;
    float4 v = make_float4(0.f, 0.f, 0.f, 0.f);
    if (row < Nn) v = in4[row * 32 + c4];
    _Float16* d = &sA[r * LDS_STRIDE + (c4 << 2)];
    d[0] = (_Float16)v.x; d[1] = (_Float16)v.y;
    d[2] = (_Float16)v.z; d[3] = (_Float16)v.w;
  }
  // ---- stage B: pre-packed transposed f16 weights -> LDS (padded stride)
  uint32_t* sB32 = (uint32_t*)sB;
  #pragma unroll
  for (int i = tid; i < 128 * 64; i += 256) {
    const int n  = i >> 6;
    const int kp = i & 63;
    sB32[n * (LDS_STRIDE / 2) + kp] = wt32[i];
  }
  __syncthreads();

  const int lane = tid & 31;
  const int wv   = tid >> 5;
  const int rt   = wv >> 1;        // row tile 0..3
  const int ch   = wv & 1;         // column half 0..1
  const int g    = lane >> 4;      // lane group for K-splitting
  const int nl   = lane & 15;

  v8f acc[4] = {};
  #pragma unroll
  for (int kc = 0; kc < 4; ++kc) {
    const v16h a = frag_ld(&sA[(rt * 16 + nl) * LDS_STRIDE + kc * 32], g);
    #pragma unroll
    for (int ct = 0; ct < 4; ++ct) {
      const v16h b = frag_ld(&sB[((ch * 4 + ct) * 16 + nl) * LDS_STRIDE + kc * 32], g);
      acc[ct] = __builtin_amdgcn_wmma_f32_16x16x32_f16(
          /*neg_a=*/false, a, /*neg_b=*/false, b,
          /*c_mod=*/(short)0, acc[ct], /*reuse_a=*/false, /*reuse_b=*/false);
    }
  }

  // ---- epilogue: bias + ReLU (+ BN scale/shift), optional dual store
  #pragma unroll
  for (int ct = 0; ct < 4; ++ct) {
    const int n  = (ch * 4 + ct) * 16 + nl;
    const float s  = bnS ? bnS[n] : 1.f;
    const float t  = bnS ? bnT[n] : 0.f;
    const float bv = bias[n];
    const int mbase = rt * 16 + (g << 3);
    #pragma unroll
    for (int v = 0; v < 8; ++v) {
      const int row = row0 + mbase + v;
      if (row < Nn) {
        const float z = fmaxf(acc[ct][v] + bv, 0.f);
        const float y = z * s + t;
        out[row * 128 + n] = y;
        if (out2) out2[row * 128 + n] = y;
      }
    }
  }
}

// ---------------------------------------------------------------------------
// GIN aggregation: out[dst,:] += h[src,:] for every edge. out pre-seeded
// with h (self term) by the preceding GEMM's dual store. 2 edges / block,
// 1 feature / thread, f32 L2 atomics (h is L2-resident: 25.6 MB << 192 MB).
// ---------------------------------------------------------------------------
__global__ __launch_bounds__(256)
void edge_aggregate(const float* __restrict__ h, float* __restrict__ out,
                    const int* __restrict__ ei, int E) {
  const int e = blockIdx.x * 2 + (threadIdx.x >> 7);
  if (e >= E) return;
  const int f   = threadIdx.x & 127;
  const int src = ei[2 * e];
  const int dst = ei[2 * e + 1];
  unsafeAtomicAdd(&out[dst * 128 + f], h[src * 128 + f]);
}

__global__ void zero_buf(float* __restrict__ o, int n) {
  const int i = blockIdx.x * 256 + threadIdx.x;
  if (i < n) o[i] = 0.f;
}

// ---------------------------------------------------------------------------
// Final graph pooling: out[batch[n],:] += h[n,:]. `batch` is sorted, so each
// thread (one per feature) walks 64 consecutive nodes with coalesced reads,
// accumulating run-lengths in a register and flushing one atomic per segment
// boundary (~30x fewer atomics than the naive form).
// ---------------------------------------------------------------------------
#define POOL_NODES 64
__global__ __launch_bounds__(128)
void pool_segments(const float* __restrict__ h, const int* __restrict__ batch,
                   float* __restrict__ out, int Nn) {
  const int f  = threadIdx.x;              // 0..127
  const int n0 = blockIdx.x * POOL_NODES;
  int   cur_g = -1;
  float acc   = 0.f;
  for (int i = 0; i < POOL_NODES; ++i) {
    const int n = n0 + i;
    if (n >= Nn) break;
    const int gid = batch[n];
    if (gid != cur_g) {
      if (cur_g >= 0) unsafeAtomicAdd(&out[cur_g * 128 + f], acc);
      cur_g = gid;
      acc   = 0.f;
    }
    acc += h[n * 128 + f];
  }
  if (cur_g >= 0) unsafeAtomicAdd(&out[cur_g * 128 + f], acc);
}

// ---------------------------------------------------------------------------
extern "C" void kernel_launch(void* const* d_in, const int* in_sizes, int n_in,
                              void* d_out, int out_size, void* d_ws, size_t ws_size,
                              hipStream_t stream) {
  const float* x     = (const float*)d_in[0];
  const int*   ei    = (const int*)  d_in[1];
  const int*   batch = (const int*)  d_in[2];
  const float* W1 = (const float*)d_in[3];  const float* b1 = (const float*)d_in[4];
  const float* g1 = (const float*)d_in[5];  const float* be1 = (const float*)d_in[6];
  const float* mm1 = (const float*)d_in[7]; const float* mv1 = (const float*)d_in[8];
  const float* W2 = (const float*)d_in[9];  const float* b2 = (const float*)d_in[10];
  const float* W3 = (const float*)d_in[11]; const float* b3 = (const float*)d_in[12];
  const float* g2 = (const float*)d_in[13]; const float* be2 = (const float*)d_in[14];
  const float* mm2 = (const float*)d_in[15]; const float* mv2 = (const float*)d_in[16];
  const float* W4 = (const float*)d_in[17]; const float* b4 = (const float*)d_in[18];

  const int N = in_sizes[0] / HID;
  const int E = in_sizes[1] / 2;

  char* ws = (char*)d_ws;
  const size_t hbytes = (size_t)N * HID * sizeof(float);
  float*     bufA = (float*)ws;
  float*     bufB = (float*)(ws + hbytes);
  _Float16*  wt   = (_Float16*)(ws + 2 * hbytes);
  float*     bn   = (float*)(ws + 2 * hbytes + 4 * 128 * 128 * sizeof(_Float16));

  pack_weights<<<256, 256, 0, stream>>>(W1, W2, W3, W4, g1, be1, mm1, mv1,
                                        g2, be2, mm2, mv2, wt, bn);

  const uint32_t* wt32 = (const uint32_t*)wt;
  const int nb = (N + 63) / 64;
  const int eb = (E + 1) / 2;

  // L1: bn1(relu(x@W1+b1)) -> bufA
  gemm128<<<nb, 256, 0, stream>>>(x, bufA, nullptr, wt32 + 0 * 8192, b1,
                                  bn, bn + 128, N);
  // L2: relu(h@W2+b2) -> bufB, dual-stored into bufA as aggregation seed
  gemm128<<<nb, 256, 0, stream>>>(bufA, bufB, bufA, wt32 + 1 * 8192, b2,
                                  nullptr, nullptr, N);
  // agg1: bufA += scatter(bufB)
  edge_aggregate<<<eb, 256, 0, stream>>>(bufB, bufA, ei, E);
  // L3: bn2(relu(a@W3+b3)) -> bufB, dual-stored into bufA as agg seed
  gemm128<<<nb, 256, 0, stream>>>(bufA, bufB, bufA, wt32 + 2 * 8192, b3,
                                  bn + 256, bn + 384, N);
  // agg2: bufA += scatter(bufB)
  edge_aggregate<<<eb, 256, 0, stream>>>(bufB, bufA, ei, E);
  // L4: relu(a@W4+b4) -> bufB
  gemm128<<<nb, 256, 0, stream>>>(bufA, bufB, nullptr, wt32 + 3 * 8192, b4,
                                  nullptr, nullptr, N);

  // graph pooling (segment-run accumulation over sorted batch)
  zero_buf<<<(out_size + 255) / 256, 256, 0, stream>>>((float*)d_out, out_size);
  pool_segments<<<(N + POOL_NODES - 1) / POOL_NODES, 128, 0, stream>>>(
      bufB, batch, (float*)d_out, N);
}
